// Transform_CA_SA_mamba_12240656793954
// MI455X (gfx1250) — matrix operand order repaired and checked by reading, not compiled, not run on hardware
//
#include <hip/hip_runtime.h>
#include <hip/hip_bf16.h>
#include <math.h>

typedef __attribute__((ext_vector_type(16))) _Float16 v16h;
typedef __attribute__((ext_vector_type(8)))  float    v8f;

// ---------------------------------------------------------------- utilities
__device__ __forceinline__ float siluf(float v) { return v / (1.0f + expf(-v)); }
__device__ __forceinline__ float softplusf(float v) { return v > 20.0f ? v : log1pf(expf(v)); }
__device__ __forceinline__ float gelu_tanh(float v) {
  float v3 = v * v * v;
  return 0.5f * v * (1.0f + tanhf(0.7978845608028654f * (v + 0.044715f * v3)));
}

// --------------------------------------------------- MVN (+nearest resize)
// Computes per-(b,ch) mean/var (ddof=1) over the RESIZED map and writes the
// normalized values transposed into x with layout (b, l, Ctot) at column coff.
__global__ __launch_bounds__(256) void mvn_kernel(
    const float* __restrict__ src, int C, int hi, int wi, int ho, int wo,
    float* __restrict__ x, int Ctot, int coff) {
  __shared__ float cache[4096];
  __shared__ float red[256];
  __shared__ float red2[256];
  int b  = blockIdx.x / C;
  int ch = blockIdx.x % C;
  int HW = ho * wo;
  const float* sp = src + (size_t)(b * C + ch) * hi * wi;
  float s = 0.f, ss = 0.f;
  for (int i = threadIdx.x; i < HW; i += blockDim.x) {
    int r = i / wo, cc = i % wo;
    int sr = (r * hi) / ho;
    int sc = (cc * wi) / wo;
    float v = sp[sr * wi + sc];
    cache[i] = v;
    s += v; ss += v * v;
  }
  red[threadIdx.x] = s; red2[threadIdx.x] = ss;
  __syncthreads();
  for (int off = 128; off >= 1; off >>= 1) {
    if ((int)threadIdx.x < off) {
      red[threadIdx.x]  += red[threadIdx.x + off];
      red2[threadIdx.x] += red2[threadIdx.x + off];
    }
    __syncthreads();
  }
  float m   = red[0] / (float)HW;
  float var = (red2[0] - (float)HW * m * m) / (float)(HW - 1);
  float inv = rsqrtf(var + 1e-5f);
  for (int i = threadIdx.x; i < HW; i += blockDim.x) {
    x[((size_t)(b * HW + i)) * Ctot + coff + ch] = (cache[i] - m) * inv;
  }
}

// ---------------------------------------------------------- FiLM modulation
// mod[b, j] = clip_s[b,:] . fs_w[j,:] + fs_b[j]   (j < 6c, K = 512)
__global__ __launch_bounds__(256) void film_kernel(
    const float* __restrict__ clip, const float* __restrict__ w,
    const float* __restrict__ bsrc, float* __restrict__ mod,
    int B, int c6, int Kc) {
  int idx = blockIdx.x * blockDim.x + threadIdx.x;
  if (idx >= B * c6) return;
  int b = idx / c6, j = idx % c6;
  const float* cr = clip + (size_t)b * Kc;
  const float* wr = w + (size_t)j * Kc;
  float acc = bsrc[j];
  for (int k = 0; k < Kc; ++k) acc += cr[k] * wr[k];
  mod[idx] = acc;
}

// --------------------------------------------------- LayerNorm + modulate
// One wave per row of length c. out = ln(x) * (1 + sc) + sh
__global__ __launch_bounds__(256) void ln_mod_kernel(
    const float* __restrict__ x, const float* __restrict__ mod,
    float* __restrict__ out, int M, int L, int c, int c6,
    int sh_off, int sc_off) {
  int wave = (blockIdx.x * blockDim.x + threadIdx.x) >> 5;
  int lane = threadIdx.x & 31;
  if (wave >= M) return;
  int b = wave / L;
  const float* row = x + (size_t)wave * c;
  float s = 0.f, ss = 0.f;
  for (int j = lane; j < c; j += 32) { float v = row[j]; s += v; ss += v * v; }
  for (int o = 16; o >= 1; o >>= 1) {
    s  += __shfl_xor(s, o, 32);
    ss += __shfl_xor(ss, o, 32);
  }
  float m   = s / (float)c;
  float var = ss / (float)c - m * m;
  float inv = rsqrtf(var + 1e-6f);
  const float* sh = mod + (size_t)b * c6 + sh_off;
  const float* sc = mod + (size_t)b * c6 + sc_off;
  float* orow = out + (size_t)wave * c;
  for (int j = lane; j < c; j += 32) {
    float v = (row[j] - m) * inv;
    orow[j] = v * (1.0f + sc[j]) + sh[j];
  }
}

// ------------------------------------------------------- WMMA f16 GEMM
// C[M,N] = A[M,K] * W[N,K]^T (+bias) (+gelu).
// Block = 8 waves computing one 16x128 strip: the 16xK A strip is converted
// f32->f16 cooperatively into LDS once and shared by all 8 waves (8x less
// global A traffic); each wave owns one 16x16 N-tile and streams
// ds_load_b128 A-fragments + K-contiguous global B rows into
// v_wmma_f32_16x16x32_f16 with f32 accumulation.
// Requires K <= 1024, K % 32 == 0, M % 16 == 0, N % 16 == 0.
template <int ACT>
__global__ __launch_bounds__(256) void gemm_wmma_kernel(
    const float* __restrict__ A, const float* __restrict__ W,
    const float* __restrict__ bias, float* __restrict__ C,
    int M, int N, int K) {
  __shared__ _Float16 As[16 * 1024];   // 32 KB (of 320 KB per WGP)
  int tm = blockIdx.x;
  int nt = N >> 4;
  // cooperative stage: A strip (16 x K) -> f16 in LDS, row-major
  const float* Abase = A + (size_t)tm * 16 * K;
  int total = 16 * K;
  for (int i = threadIdx.x; i < total; i += 256) {
    As[i] = (_Float16)Abase[i];
  }
  __syncthreads();
  int wid  = threadIdx.x >> 5;
  int lane = threadIdx.x & 31;
  int tn = blockIdx.y * 8 + wid;
  if (tn >= nt) return;               // whole wave exits; EXEC stays full
  int hl = lane >> 4, l16 = lane & 15;
  const float* Wrow = W + (size_t)(tn * 16 + l16) * K;
  const _Float16* Alane = As + l16 * K;
  v8f acc = {};
  for (int k0 = 0; k0 < K; k0 += 32) {
    v16h av, bv;
    // A 16x32 lane layout: lanes<16 hold K 0..7 / 16..23; lanes>=16 8..15 / 24..31
    const _Float16* ap = Alane + k0 + hl * 8;
#pragma unroll
    for (int i = 0; i < 8; ++i) {
      av[i]     = ap[i];
      av[i + 8] = ap[i + 16];
    }
    // B 32x16: lane n holds column n, K 0..7 / 16..23 (lanes<16) or 8..15 / 24..31 (lanes>=16)
    const float* bp = Wrow + k0 + hl * 8;
#pragma unroll
    for (int i = 0; i < 8; ++i) {
      bv[i]     = (_Float16)bp[i];
      bv[i + 8] = (_Float16)bp[i + 16];
    }
    acc = __builtin_amdgcn_wmma_f32_16x16x32_f16(
        false, av, false, bv, (short)0, acc, false, false);
  }
  int col = tn * 16 + l16;
  float bb = bias ? bias[col] : 0.0f;
#pragma unroll
  for (int r = 0; r < 8; ++r) {
    int row = tm * 16 + hl * 8 + r;   // C/D layout: lanes>=16 carry M+8
    float v = acc[r] + bb;
    if (ACT == 1) v = gelu_tanh(v);
    C[(size_t)row * N + col] = v;
  }
}

// ---------------------------------------------- causal depthwise conv + SiLU
__global__ __launch_bounds__(256) void conv_silu_kernel(
    const float* __restrict__ xz, const float* __restrict__ cw,
    const float* __restrict__ cb, float* __restrict__ xc,
    int M, int L, int din, int c4) {
  int idx = blockIdx.x * blockDim.x + threadIdx.x;
  if (idx >= M * din) return;
  int row = idx / din, ch = idx % din;
  int b = row / L, l = row % L;
  const float* w = cw + (size_t)ch * 4;
  float acc = cb[ch];
#pragma unroll
  for (int t = 0; t < 4; ++t) {
    int lp = l - 3 + t;
    if (lp >= 0) acc += w[t] * xz[((size_t)(b * L + lp)) * c4 + ch];
  }
  xc[(size_t)row * din + ch] = siluf(acc);
}

// ------------------------------------------- x_proj: dbl = xc @ x_proj_w^T
__global__ __launch_bounds__(256) void dbl_kernel(
    const float* __restrict__ xc, const float* __restrict__ xw,
    float* __restrict__ dbl, int M, int din, int nd) {
  int idx = blockIdx.x * blockDim.x + threadIdx.x;
  if (idx >= M * nd) return;
  int row = idx / nd, j = idx % nd;
  const float* a = xc + (size_t)row * din;
  const float* w = xw + (size_t)j * din;
  float acc = 0.f;
  for (int k = 0; k < din; ++k) acc += a[k] * w[k];
  dbl[(size_t)row * nd + j] = acc;
}

// --------------------------- delta = softplus(dt @ dt_proj_w^T + dt_bias)
__global__ __launch_bounds__(256) void delta_kernel(
    const float* __restrict__ dbl, const float* __restrict__ dtw,
    const float* __restrict__ dtb, float* __restrict__ delta,
    int M, int din, int nd, int dtr) {
  int idx = blockIdx.x * blockDim.x + threadIdx.x;
  if (idx >= M * din) return;
  int row = idx / din, ch = idx % din;
  const float* a = dbl + (size_t)row * nd;
  const float* w = dtw + (size_t)ch * dtr;
  float acc = dtb[ch];
  for (int t = 0; t < dtr; ++t) acc += a[t] * w[t];
  delta[(size_t)row * din + ch] = softplusf(acc);
}

// --------------------------------------------------- selective scan (SSM)
// One wave = 2 channels x 16 states. Sequential over L; 16-lane xor-reduce
// of h*C; fused epilogue y = (sum + xc*D) * silu(z).
__global__ __launch_bounds__(256) void scan_kernel(
    const float* __restrict__ delta, const float* __restrict__ xc,
    const float* __restrict__ dbl, const float* __restrict__ xz,
    const float* __restrict__ A_log, const float* __restrict__ D,
    float* __restrict__ y,
    int B, int L, int c, int din, int c4, int nd, int dtr) {
  int gtid = blockIdx.x * blockDim.x + threadIdx.x;
  int wave = gtid >> 5;
  if (wave >= B * c) return;
  int lane = threadIdx.x & 31;
  int b = wave / c, chp = wave % c;
  int hl = lane >> 4, s = lane & 15;
  int ch = chp * 2 + hl;
  float Av = -expf(A_log[(size_t)ch * 16 + s]);
  float Dv = D[ch];
  float h = 0.f;
  size_t rowbase = (size_t)b * L;
  for (int l = 0; l < L; ++l) {
    size_t row = rowbase + l;
    float dv = delta[row * din + ch];
    float xv = xc[row * din + ch];
    float Bm = dbl[row * nd + dtr + s];
    float Cm = dbl[row * nd + dtr + 16 + s];
    h = expf(dv * Av) * h + dv * Bm * xv;
    float p = h * Cm;
    p += __shfl_xor(p, 1, 16);
    p += __shfl_xor(p, 2, 16);
    p += __shfl_xor(p, 4, 16);
    p += __shfl_xor(p, 8, 16);
    if (s == 0) {
      float zv = xz[row * c4 + din + ch];   // z = xz[:, 2c + ch]
      y[row * din + ch] = (p + xv * Dv) * siluf(zv);
    }
  }
}

// --------------------------------------------------- x += g[b,:] * u
__global__ __launch_bounds__(256) void gain_add_kernel(
    float* __restrict__ x, const float* __restrict__ u,
    const float* __restrict__ mod, int M, int L, int c, int c6, int goff) {
  int idx = blockIdx.x * blockDim.x + threadIdx.x;
  if (idx >= M * c) return;
  int row = idx / c, j = idx % c;
  int b = row / L;
  x[idx] += mod[(size_t)b * c6 + goff + j] * u[idx];
}

// --------------------------- (b,l,o) -> (b,o,h,w) transpose + prep bias
__global__ __launch_bounds__(256) void prep_store_kernel(
    const float* __restrict__ ptmp, const float* __restrict__ pb,
    float* __restrict__ out, int B, int L, int o) {
  int idx = blockIdx.x * blockDim.x + threadIdx.x;
  if (idx >= B * o * L) return;
  int b = idx / (o * L);
  int rem = idx % (o * L);
  int oc = rem / L, l = rem % L;
  out[idx] = ptmp[((size_t)b * L + l) * o + oc] + pb[oc];
}

// ================================================================ host side
struct Params {
  const float *fs_w, *fs_b, *in_proj_w, *conv_w, *conv_b, *x_proj_w,
              *dt_proj_w, *dt_bias, *A_log, *D, *out_proj_w,
              *fc1_w, *fc1_b, *fc2_w, *fc2_b, *prep_w, *prep_b;
};

static Params get_params(void* const* d_in, int s) {
  Params p;
  const float** f = (const float**)&p;
  int base = 5 + 17 * s;
  for (int i = 0; i < 17; ++i) f[i] = (const float*)d_in[base + i];
  return p;
}

static inline void launch_gemm(int act, const float* A, const float* W,
                               const float* bias, float* C,
                               int M, int N, int K, hipStream_t s) {
  dim3 grid(M / 16, (N / 16 + 7) / 8);
  if (act)
    gemm_wmma_kernel<1><<<grid, 256, 0, s>>>(A, W, bias, C, M, N, K);
  else
    gemm_wmma_kernel<0><<<grid, 256, 0, s>>>(A, W, bias, C, M, N, K);
}

static void run_scale(hipStream_t stream, const Params& P, const float* clip,
                      float* x, float* xmod, float* xz, float* xc,
                      float* delta, float* y, float* dbl, float* mod,
                      float* out_seg, int c, int o, int L) {
  const int B = 4;
  const int M = B * L;
  const int din = 2 * c, c4 = 4 * c, c6 = 6 * c;
  const int dtr = (c + 15) / 16, nd = dtr + 32;

  // FiLM modulation vector
  { int n = B * c6;
    film_kernel<<<(n + 255) / 256, 256, 0, stream>>>(clip, P.fs_w, P.fs_b, mod, B, c6, 512); }
  // attn branch: ln + modulate
  { int blocks = (M * 32 + 255) / 256;
    ln_mod_kernel<<<blocks, 256, 0, stream>>>(x, mod, xmod, M, L, c, c6, 0, c); }
  // in_proj (WMMA)
  launch_gemm(0, xmod, P.in_proj_w, nullptr, xz, M, c4, c, stream);
  // causal depthwise conv + silu
  { int n = M * din;
    conv_silu_kernel<<<(n + 255) / 256, 256, 0, stream>>>(xz, P.conv_w, P.conv_b, xc, M, L, din, c4); }
  // x_proj (narrow N -> VALU dots)
  { int n = M * nd;
    dbl_kernel<<<(n + 255) / 256, 256, 0, stream>>>(xc, P.x_proj_w, dbl, M, din, nd); }
  // dt_proj + softplus
  { int n = M * din;
    delta_kernel<<<(n + 255) / 256, 256, 0, stream>>>(dbl, P.dt_proj_w, P.dt_bias, delta, M, din, nd, dtr); }
  // selective scan, fused with D-skip and silu(z) gate
  { int waves = B * c;
    int blocks = (waves + 7) / 8;
    scan_kernel<<<blocks, 256, 0, stream>>>(delta, xc, dbl, xz, P.A_log, P.D, y,
                                            B, L, c, din, c4, nd, dtr); }
  // out_proj (WMMA) -> reuse xmod as mamba output
  launch_gemm(0, y, P.out_proj_w, nullptr, xmod, M, c, din, stream);
  // x += g_a * mamba_out
  { int n = M * c;
    gain_add_kernel<<<(n + 255) / 256, 256, 0, stream>>>(x, xmod, mod, M, L, c, c6, 2 * c); }
  // mlp branch: ln + modulate -> xmod
  { int blocks = (M * 32 + 255) / 256;
    ln_mod_kernel<<<blocks, 256, 0, stream>>>(x, mod, xmod, M, L, c, c6, 3 * c, 4 * c); }
  // fc1 + gelu (WMMA) -> y (reused)
  launch_gemm(1, xmod, P.fc1_w, P.fc1_b, y, M, c, c, stream);
  // fc2 (WMMA) -> xc (reused)
  launch_gemm(0, y, P.fc2_w, P.fc2_b, xc, M, c, c, stream);
  // x += g_m * u
  { int n = M * c;
    gain_add_kernel<<<(n + 255) / 256, 256, 0, stream>>>(x, xc, mod, M, L, c, c6, 5 * c); }
  // prep 1x1 conv (WMMA) -> delta buffer (reused, M*2c >= M*o)
  launch_gemm(0, x, P.prep_w, nullptr, delta, M, o, c, stream);
  // transpose to (b,o,h,w) + bias, into output segment
  { int n = B * o * L;
    prep_store_kernel<<<(n + 255) / 256, 256, 0, stream>>>(delta, P.prep_b, out_seg, B, L, o); }
}

extern "C" void kernel_launch(void* const* d_in, const int* in_sizes, int n_in,
                              void* d_out, int out_size, void* d_ws, size_t ws_size,
                              hipStream_t stream) {
  (void)in_sizes; (void)n_in; (void)out_size; (void)ws_size;

  const float* F_clip_s = (const float*)d_in[1];   // (4,1,512) -> (4,512)
  const float* F_c0 = (const float*)d_in[2];       // (4,64,64,64)
  const float* F_c1 = (const float*)d_in[3];       // (4,128,32,32)
  const float* F_c2 = (const float*)d_in[4];       // (4,320,16,16)
  float* out = (float*)d_out;

  Params P0 = get_params(d_in, 0);
  Params P1 = get_params(d_in, 1);
  Params P2 = get_params(d_in, 2);

  // workspace layout (floats), sized for the largest scale, reused serially
  const size_t MAXX   = (size_t)16384 * 64;    // x / xmod: M*c
  const size_t MAXXZ  = (size_t)16384 * 256;   // xz: M*4c
  const size_t MAXXC  = (size_t)16384 * 128;   // xc / delta / y: M*2c
  const size_t MAXDBL = (size_t)16384 * 64;    // dbl: M*(dtr+32)
  float* ws      = (float*)d_ws;
  float* B_x     = ws;
  float* B_xmod  = B_x + MAXX;
  float* B_xz    = B_xmod + MAXX;
  float* B_xc    = B_xz + MAXXZ;
  float* B_delta = B_xc + MAXXC;
  float* B_y     = B_delta + MAXXC;
  float* B_dbl   = B_y + MAXXC;
  float* B_mod   = B_dbl + MAXDBL;

  // ---- scale 0: k0 = mvn(F_c0), c=64, o=64, L=4096
  mvn_kernel<<<4 * 64, 256, 0, stream>>>(F_c0, 64, 64, 64, 64, 64, B_x, 64, 0);
  run_scale(stream, P0, F_clip_s, B_x, B_xmod, B_xz, B_xc, B_delta, B_y, B_dbl,
            B_mod, out, 64, 64, 4096);

  // ---- scale 1: k1 = [mvn(resize(F_c0,32)), mvn(F_c1)], c=192, o=128, L=1024
  mvn_kernel<<<4 * 64, 256, 0, stream>>>(F_c0, 64, 64, 64, 32, 32, B_x, 192, 0);
  mvn_kernel<<<4 * 128, 256, 0, stream>>>(F_c1, 128, 32, 32, 32, 32, B_x, 192, 64);
  run_scale(stream, P1, F_clip_s, B_x, B_xmod, B_xz, B_xc, B_delta, B_y, B_dbl,
            B_mod, out + (size_t)4 * 64 * 4096, 192, 128, 1024);

  // ---- scale 2: k2 = [mvn(rs(F_c0,16)), mvn(rs(F_c1,16)), mvn(F_c2)], c=512, o=320, L=256
  mvn_kernel<<<4 * 64, 256, 0, stream>>>(F_c0, 64, 64, 64, 16, 16, B_x, 512, 0);
  mvn_kernel<<<4 * 128, 256, 0, stream>>>(F_c1, 128, 32, 32, 16, 16, B_x, 512, 64);
  mvn_kernel<<<4 * 320, 256, 0, stream>>>(F_c2, 320, 16, 16, 16, 16, B_x, 512, 192);
  run_scale(stream, P2, F_clip_s, B_x, B_xmod, B_xz, B_xc, B_delta, B_y, B_dbl,
            B_mod, out + (size_t)4 * 64 * 4096 + (size_t)4 * 128 * 1024,
            512, 320, 256);
}
